// biLSTM_88450556494553
// MI455X (gfx1250) — matrix operand order repaired and checked by reading, not compiled, not run on hardware
//
#include <hip/hip_runtime.h>

// ---------------------------------------------------------------------------
// Bidirectional LSTM on MI455X (gfx1250), bf16 WMMA path, v4.
// B=64, T=512, E=512, H=1024, NC=1024. Gates stacked g,i,f,o.
//
// v4 changes vs v3:
//  * K-loop split into emb-phase (tiles 0..15) and h-phase (tiles 16..47):
//    removes the per-tile v_cmp/v_cndmask B-pointer select.
//  * All fragment loads driven by running pointers (+32 elems/tile) so the
//    per-iteration address math collapses to one add per stream.
//  * Ping-pong double buffering + partial-wait WMMA overlap retained.
// ---------------------------------------------------------------------------

typedef __bf16 bf16;
typedef __attribute__((ext_vector_type(16))) __bf16 v16bf;
typedef __attribute__((ext_vector_type(8)))  __bf16 v8bf;
typedef __attribute__((ext_vector_type(8)))  float  v8f;

#define SHUF16(lo, hi) __builtin_shufflevector(lo, hi, 0,1,2,3,4,5,6,7,8,9,10,11,12,13,14,15)

static constexpr int Bsz  = 64;
static constexpr int Tlen = 512;
static constexpr int Edim = 512;
static constexpr int Hdim = 1024;
static constexpr int NC   = 1024;
static constexpr int KTOT = Edim + Hdim;   // 1536
static constexpr int KT2  = 2 * Hdim;      // 2048 (projection K)

// A fragment (16x32 bf16): per lane 16 contiguous elements.
__device__ __forceinline__ v16bf load_fragA(const bf16* p) {
    v8bf lo = *(const v8bf*)p;
    v8bf hi = *(const v8bf*)(p + 8);
    return SHUF16(lo, hi);
}
// B fragment (32x16 bf16): per lane 16 contiguous elements.
__device__ __forceinline__ v16bf load_fragB(const bf16* p) {
    v8bf lo = *(const v8bf*)p;
    v8bf hi = *(const v8bf*)(p + 8);
    return SHUF16(lo, hi);
}

// ---------------------------------------------------------------- prep ----

__global__ void f2bf_kernel(const float* __restrict__ src, bf16* __restrict__ dst, int n) {
    int i = blockIdx.x * blockDim.x + threadIdx.x;
    if (i < n) dst[i] = (bf16)src[i];
}

// Wc[r][k] = k<E ? Wx[r][k] : Wh[r][k-E], bf16, r in [0,4H)
__global__ void build_wc_kernel(const float* __restrict__ Wx, const float* __restrict__ Wh,
                                bf16* __restrict__ Wc) {
    int i = blockIdx.x * blockDim.x + threadIdx.x;
    if (i < 4 * Hdim * KTOT) {
        int r = i / KTOT, k = i % KTOT;
        float v = (k < Edim) ? Wx[(size_t)r * Edim + k] : Wh[(size_t)r * Hdim + (k - Edim)];
        Wc[i] = (bf16)v;
    }
}

__global__ void init_state_kernel(bf16* __restrict__ h0, bf16* __restrict__ h1,
                                  float* __restrict__ c) {
    int i = blockIdx.x * blockDim.x + threadIdx.x;
    if (i < 2 * Bsz * Hdim) { h0[i] = (bf16)0.0f; h1[i] = (bf16)0.0f; c[i] = 0.0f; }
}

// ------------------------------------------------------------- LSTM step ----
// grid = (H/16 m-tiles, 1, 2 directions), block = 64 (2 waves).
// Wave w owns batch half [32w, 32w+32): 2 n-tiles x 4 gates = 8 accums.
// K-loop: ping-pong buffers, running pointers, emb-phase then h-phase.

__global__ __launch_bounds__(64)
void lstm_step_kernel(const bf16* __restrict__ Wc_f, const bf16* __restrict__ Wc_b,
                      const bf16* __restrict__ embb, const int* __restrict__ xIdx,
                      const bf16* __restrict__ hin,  bf16* __restrict__ hout,
                      float* __restrict__ cbuf,
                      const float* __restrict__ b_f, const float* __restrict__ b_b,
                      int s) {
    const int dir  = blockIdx.z;
    const int t    = dir ? (Tlen - 1 - s) : s;
    const bf16*  Wc   = dir ? Wc_b : Wc_f;
    const float* bias = dir ? b_b : b_f;

    const int lane = threadIdx.x & 31;
    const int wv   = threadIdx.x >> 5;     // wave id: batch half
    const int kh   = lane >> 4;            // half-wave 0/1
    const int l15  = lane & 15;
    const int m0   = blockIdx.x * 16;      // hidden-tile base

    const int aRow = m0 + l15;             // A row within each gate block
    const int kbA  = kh * 16;              // A frag K sub-offset
    const int kbB  = kh * 16;              // B frag K sub-offset

    // Running load pointers: advance 32 elems per K-tile.
    const bf16* aP[4];
    #pragma unroll
    for (int g = 0; g < 4; ++g)
        aP[g] = Wc + ((size_t)(g * Hdim + aRow)) * KTOT + kbA;

    int n_[2];
    const bf16* bP[2];       // current B pointer (emb phase first)
    const bf16* hRowP[2];    // h-phase base
    #pragma unroll
    for (int j = 0; j < 2; ++j) {
        const int n = wv * 32 + j * 16 + l15;
        n_[j] = n;
        const int idx = xIdx[(size_t)n * Tlen + t];
        bP[j]    = embb + (size_t)idx * Edim + kbB;
        hRowP[j] = hin + ((size_t)dir * Bsz + n) * Hdim + kbB;
    }

    v8f acc[4][2] = {};
    v16bf a0[4], a1[4], b0[2], b1[2];

    // Load one K-tile into (af,bf_) and advance all running pointers.
    auto loadTo = [&](v16bf (&af)[4], v16bf (&bf_)[2]) {
        #pragma unroll
        for (int g = 0; g < 4; ++g) { af[g] = load_fragA(aP[g]); aP[g] += 32; }
        #pragma unroll
        for (int j = 0; j < 2; ++j) { bf_[j] = load_fragB(bP[j]); bP[j] += 32; }
    };
    auto mmaTile = [&](v16bf (&af)[4], v16bf (&bf_)[2]) {
        #pragma unroll
        for (int g = 0; g < 4; ++g) {
            #pragma unroll
            for (int j = 0; j < 2; ++j) {
                acc[g][j] = __builtin_amdgcn_wmma_f32_16x16x32_bf16(
                    false, af[g], false, bf_[j], (short)0, acc[g][j], false, false);
            }
        }
    };

    constexpr int NE = Edim / 32;  // 16 emb-phase tiles
    constexpr int NH = Hdim / 32;  // 32 h-phase tiles

    loadTo(a0, b0);                        // tile 0 (emb)
    loadTo(a1, b1);                        // tile 1 (emb)
    for (int i = 0; i < (NE - 2) / 2; ++i) {   // mma tiles 0..13, load 2..15
        __builtin_prefetch(aP[0] + 256, 0, 1); // weight stream, 8 tiles ahead
        mmaTile(a0, b0); loadTo(a0, b0);
        mmaTile(a1, b1); loadTo(a1, b1);
    }
    #pragma unroll
    for (int j = 0; j < 2; ++j) bP[j] = hRowP[j];  // switch B stream to h
    for (int i = 0; i < NH / 2; ++i) {         // mma tiles 14..45, load 16..47
        __builtin_prefetch(aP[0] + 256, 0, 1);
        mmaTile(a0, b0); loadTo(a0, b0);
        mmaTile(a1, b1); loadTo(a1, b1);
    }
    mmaTile(a0, b0);   // tile 46
    mmaTile(a1, b1);   // tile 47

    // ---- fused gate epilogue: lane owns (n_[j], m0 + 8*kh + r), r=0..7
    const int mrow0 = m0 + kh * 8;
    #pragma unroll
    for (int j = 0; j < 2; ++j) {
        float* cRow    = cbuf + ((size_t)dir * Bsz + n_[j]) * Hdim + mrow0;
        bf16*  hOutRow = hout + ((size_t)dir * Bsz + n_[j]) * Hdim + mrow0;
        #pragma unroll
        for (int r = 0; r < 8; ++r) {
            const int m = mrow0 + r;
            const float zg = acc[0][j][r] + bias[0 * Hdim + m];
            const float zi = acc[1][j][r] + bias[1 * Hdim + m];
            const float zf = acc[2][j][r] + bias[2 * Hdim + m];
            const float zo = acc[3][j][r] + bias[3 * Hdim + m];
            const float g = tanhf(zg);
            const float i = 1.0f / (1.0f + __expf(-zi));
            const float f = 1.0f / (1.0f + __expf(-zf));
            const float o = 1.0f / (1.0f + __expf(-zo));
            const float c = g * i + cRow[r] * f;
            cRow[r]    = c;
            hOutRow[r] = (bf16)(tanhf(c) * o);
        }
    }
}

// ------------------------------------------------------------ projection ----
// p[b][cls] = W_ph @ [h_fwd; h_bwd] + b_p.  grid = (NC/16, B/16), wave/block.
// Same two-phase (hf then hb) ping-pong pipeline with running pointers.

__global__ __launch_bounds__(32)
void proj_kernel(const bf16* __restrict__ Wph, const bf16* __restrict__ hfin,
                 const float* __restrict__ b_p, float* __restrict__ p_ws) {
    const int lane = threadIdx.x;
    const int kh   = lane >> 4;
    const int l15  = lane & 15;
    const int m0   = blockIdx.x * 16;   // class tile
    const int n0   = blockIdx.y * 16;   // batch tile
    const int n    = n0 + l15;
    const int aRow = m0 + l15;

    const bf16* aP = Wph + (size_t)aRow * KT2 + kh * 16;
    const bf16* bP = hfin + (size_t)n * Hdim + kh * 16;             // hf phase
    const bf16* hbP = hfin + ((size_t)Bsz + n) * Hdim + kh * 16;    // hb phase

    v8f acc = {};
    v16bf a0, a1, b0, b1;
    auto loadTo = [&](v16bf& af, v16bf& bf_) {
        af = load_fragA(aP); aP += 32;
        bf_ = load_fragB(bP); bP += 32;
    };
    auto mma1 = [&](v16bf& af, v16bf& bf_) {
        acc = __builtin_amdgcn_wmma_f32_16x16x32_bf16(
            false, af, false, bf_, (short)0, acc, false, false);
    };

    constexpr int NPH = Hdim / 32;  // 32 tiles per phase
    loadTo(a0, b0);
    loadTo(a1, b1);
    for (int i = 0; i < (NPH - 2) / 2; ++i) {   // mma 0..29, load 2..31
        mma1(a0, b0); loadTo(a0, b0);
        mma1(a1, b1); loadTo(a1, b1);
    }
    bP = hbP;                                   // switch to h_bwd rows
    for (int i = 0; i < NPH / 2; ++i) {         // mma 30..61, load 32..63
        mma1(a0, b0); loadTo(a0, b0);
        mma1(a1, b1); loadTo(a1, b1);
    }
    mma1(a0, b0);   // tile 62
    mma1(a1, b1);   // tile 63

    const int mrow0 = m0 + kh * 8;
    #pragma unroll
    for (int r = 0; r < 8; ++r) {
        const int cls = mrow0 + r;
        p_ws[(size_t)n * NC + cls] = acc[r] + b_p[cls];
    }
}

// ------------------------------------------------------------ log-softmax ----

__global__ __launch_bounds__(256)
void logsoftmax_kernel(const float* __restrict__ p_ws, float* __restrict__ out) {
    __shared__ float red[256];
    const int b   = blockIdx.x;
    const int tid = threadIdx.x;
    const float* row = p_ws + (size_t)b * NC;

    float mx = -3.402823466e38f;
    for (int c = tid; c < NC; c += 256) mx = fmaxf(mx, row[c]);
    red[tid] = mx; __syncthreads();
    for (int w = 128; w > 0; w >>= 1) {
        if (tid < w) red[tid] = fmaxf(red[tid], red[tid + w]);
        __syncthreads();
    }
    const float m = red[0]; __syncthreads();

    float s = 0.0f;
    for (int c = tid; c < NC; c += 256) s += __expf(row[c] - m);
    red[tid] = s; __syncthreads();
    for (int w = 128; w > 0; w >>= 1) {
        if (tid < w) red[tid] += red[tid + w];
        __syncthreads();
    }
    const float lse = m + logf(red[0]);

    for (int c = tid; c < NC; c += 256) out[(size_t)b * NC + c] = row[c] - lse;
}

// ---------------------------------------------------------------- launch ----

extern "C" void kernel_launch(void* const* d_in, const int* in_sizes, int n_in,
                              void* d_out, int out_size, void* d_ws, size_t ws_size,
                              hipStream_t stream) {
    (void)in_sizes; (void)n_in; (void)out_size; (void)ws_size;

    const int*   x    = (const int*)  d_in[0];   // (B, T)
    const float* emb  = (const float*)d_in[1];   // (NC, E)
    const float* Wx_f = (const float*)d_in[2];   // (4H, E)
    const float* Wh_f = (const float*)d_in[3];   // (4H, H)
    const float* b_f  = (const float*)d_in[4];   // (4H, 1)
    const float* Wx_b = (const float*)d_in[5];
    const float* Wh_b = (const float*)d_in[6];
    const float* b_b  = (const float*)d_in[7];
    const float* W_ph = (const float*)d_in[8];   // (NC, 2H)
    const float* b_p  = (const float*)d_in[9];   // (NC, 1)
    float* out = (float*)d_out;                  // (B, NC)

    // workspace layout (bytes, all 256-aligned)
    char* ws = (char*)d_ws;
    bf16*  emb_bf = (bf16*)(ws + 0);                         //  1,048,576 B
    bf16*  Wc_f   = (bf16*)(ws + (1u << 20));                // 12,582,912 B
    bf16*  Wc_b   = (bf16*)(ws + 13631488u);                 // 12,582,912 B
    bf16*  Wph_bf = (bf16*)(ws + 26214400u);                 //  4,194,304 B
    bf16*  hbuf0  = (bf16*)(ws + 30408704u);                 //    262,144 B
    bf16*  hbuf1  = (bf16*)(ws + 30670848u);                 //    262,144 B
    float* cbuf   = (float*)(ws + 30932992u);                //    524,288 B
    float* p_ws   = (float*)(ws + 31457280u);                //    262,144 B

    // --- prep: bf16 conversions + fused [Wx|Wh] weight assembly + state init
    {
        int n = NC * Edim;
        f2bf_kernel<<<(n + 255) / 256, 256, 0, stream>>>(emb, emb_bf, n);
    }
    {
        int n = 4 * Hdim * KTOT;
        build_wc_kernel<<<(n + 255) / 256, 256, 0, stream>>>(Wx_f, Wh_f, Wc_f);
        build_wc_kernel<<<(n + 255) / 256, 256, 0, stream>>>(Wx_b, Wh_b, Wc_b);
    }
    {
        int n = NC * KT2;
        f2bf_kernel<<<(n + 255) / 256, 256, 0, stream>>>(W_ph, Wph_bf, n);
    }
    {
        int n = 2 * Bsz * Hdim;
        init_state_kernel<<<(n + 255) / 256, 256, 0, stream>>>(hbuf0, hbuf1, cbuf);
    }

    // --- 512 sequential LSTM steps, both directions per launch, h ping-pong
    dim3 sgrid(Hdim / 16, 1, 2);
    for (int s = 0; s < Tlen; ++s) {
        bf16* hin  = (s & 1) ? hbuf1 : hbuf0;
        bf16* hout = (s & 1) ? hbuf0 : hbuf1;
        lstm_step_kernel<<<sgrid, 64, 0, stream>>>(Wc_f, Wc_b, emb_bf, x,
                                                   hin, hout, cbuf, b_f, b_b, s);
    }
    bf16* hfinal = hbuf0;  // step 511 writes hbuf0

    // --- projection + log-softmax
    dim3 pgrid(NC / 16, Bsz / 16);
    proj_kernel<<<pgrid, 32, 0, stream>>>(Wph_bf, hfinal, b_p, p_ws);
    logsoftmax_kernel<<<Bsz, 256, 0, stream>>>(p_ws, out);
}
